// MultiHeadAttention_59476707115068
// MI455X (gfx1250) — matrix operand order repaired and checked by reading, not compiled, not run on hardware
//
#include <hip/hip_runtime.h>
#include <hip/hip_bf16.h>

// ---------- CDNA5 WMMA types ----------
typedef __bf16        v16bf __attribute__((ext_vector_type(16)));
typedef float         v8f   __attribute__((ext_vector_type(8)));
typedef unsigned int  v4u   __attribute__((ext_vector_type(4)));
typedef float         v4f   __attribute__((ext_vector_type(4)));

union FragU { v4u u[2]; v16bf v; };

__device__ __forceinline__ unsigned short f32_to_bf16(float f) {
  unsigned int u = __float_as_uint(f);
  u += 0x7fffu + ((u >> 16) & 1u);           // round-to-nearest-even
  return (unsigned short)(u >> 16);
}
__device__ __forceinline__ unsigned int pack2(float a, float b) {
  return (unsigned int)f32_to_bf16(a) | ((unsigned int)f32_to_bf16(b) << 16);
}

__device__ __forceinline__ v8f wmma_bf16(v16bf a, v16bf b, v8f c) {
  // D = A(16x32 bf16) x B(32x16 bf16) + C(16x16 f32)
  return __builtin_amdgcn_wmma_f32_16x16x32_bf16(false, a, false, b, (short)0, c,
                                                 false, false);
}

// LDS byte offset for async-to-LDS VDST operand (flat aperture keeps the
// LDS offset in addr[31:0]).
__device__ __forceinline__ unsigned lds_off32(const void* p) {
  return (unsigned)(unsigned long long)p;
}
__device__ __forceinline__ void wait_async0() {
#if __has_builtin(__builtin_amdgcn_s_wait_asynccnt)
  __builtin_amdgcn_s_wait_asynccnt(0);
#else
  asm volatile("s_wait_asynccnt 0" ::: "memory");
#endif
}
// 64B (4x16B) async copy global->LDS; INST_OFFSET advances both addresses.
__device__ __forceinline__ void async_copy64(unsigned lds_dst, const void* gsrc) {
  asm volatile("global_load_async_to_lds_b128 %0, %1, off\n\t"
               "global_load_async_to_lds_b128 %0, %1, off offset:16\n\t"
               "global_load_async_to_lds_b128 %0, %1, off offset:32\n\t"
               "global_load_async_to_lds_b128 %0, %1, off offset:48"
               :: "v"(lds_dst), "v"(gsrc) : "memory");
}
// 32B (2x16B) async copy global->LDS.
__device__ __forceinline__ void async_copy32(unsigned lds_dst, const void* gsrc) {
  asm volatile("global_load_async_to_lds_b128 %0, %1, off\n\t"
               "global_load_async_to_lds_b128 %0, %1, off offset:16"
               :: "v"(lds_dst), "v"(gsrc) : "memory");
}

// A-operand fragment: rows striped over lanes (lane&15), K chunks per ISA layout:
// elems 0..7 -> K = 8*(lane>>4)+e ; elems 8..15 -> K = 16+8*(lane>>4)+(e-8)
__device__ __forceinline__ v16bf load_frag_a(const unsigned short* base, int lda,
                                             int row0, int lane) {
  const int lo = lane & 15, hi = lane >> 4;
  const unsigned short* r = base + (size_t)(row0 + lo) * lda;
  FragU f;
  f.u[0] = *(const v4u*)(r + hi * 8);
  f.u[1] = *(const v4u*)(r + 16 + hi * 8);
  return f.v;
}

// B-operand fragment from operand-major storage Bst[n][k]:
// column n = lane&15 ; elems 0..7 -> K = 8*(lane>>4)+e ; elems 8..15 -> K = 16+8*(lane>>4)+(e-8)
__device__ __forceinline__ v16bf load_frag_b(const unsigned short* base, int ldb,
                                             int col0, int koff, int lane) {
  const int lo = lane & 15, hi = lane >> 4;
  const unsigned short* r = base + (size_t)(col0 + lo) * ldb + koff + hi * 8;
  FragU f;
  f.u[0] = *(const v4u*)(r);
  f.u[1] = *(const v4u*)(r + 16);
  return f.v;
}

// ======================================================================
// Kernel 1: qkv = x @ w_qkv + b_qkv, scattered to Q/K/V bf16 [B,H,S,64]
// M = B*S = 8192, N = 3072, K = 1024.  Block tile 128x128x32, 8 waves.
// (Both operands need f32->bf16 conversion -> VALU staging path.)
// ======================================================================
__global__ __launch_bounds__(256) void qkv_gemm(const float* __restrict__ X,
                                                const float* __restrict__ W,
                                                const float* __restrict__ bias,
                                                unsigned short* __restrict__ Qb,
                                                unsigned short* __restrict__ Kb,
                                                unsigned short* __restrict__ Vb) {
  __shared__ __align__(16) unsigned short As[128][40];
  __shared__ __align__(16) unsigned short Bs[128][40];

  const int nBase = blockIdx.x * 128;
  const int mBase = blockIdx.y * 128;
  const int t = threadIdx.x;
  const int lane = t & 31;
  const int wave = t >> 5;
  const int wm = wave >> 1, wn = wave & 1;   // 4x2 wave grid -> 32x64 per wave
  const int lo = lane & 15, hi = lane >> 4;

  v8f acc[2][4] = {};

  const int ar = t >> 1, ac = (t & 1) * 16;  // A loader: 2 thr/row, 16 f32 each
  const int bk = t >> 3, bn = (t & 7) * 16;  // B loader: 8 thr/row

  for (int k0 = 0; k0 < 1024; k0 += 32) {
    {  // stage A (f32 -> bf16), row-major As[m][k]
      const v4f* s4 = (const v4f*)(X + (size_t)(mBase + ar) * 1024 + k0 + ac);
      v4f q0 = s4[0], q1 = s4[1], q2 = s4[2], q3 = s4[3];
      v4u u0, u1;
      u0.x = pack2(q0.x, q0.y); u0.y = pack2(q0.z, q0.w);
      u0.z = pack2(q1.x, q1.y); u0.w = pack2(q1.z, q1.w);
      u1.x = pack2(q2.x, q2.y); u1.y = pack2(q2.z, q2.w);
      u1.z = pack2(q3.x, q3.y); u1.w = pack2(q3.z, q3.w);
      *(v4u*)&As[ar][ac] = u0;
      *(v4u*)&As[ar][ac + 8] = u1;
    }
    {  // stage B transposed (operand-major Bs[n][k])
      const v4f* s4 = (const v4f*)(W + (size_t)(k0 + bk) * 3072 + nBase + bn);
      v4f q0 = s4[0], q1 = s4[1], q2 = s4[2], q3 = s4[3];
      float fv[16] __attribute__((aligned(16)));
      *(v4f*)&fv[0] = q0; *(v4f*)&fv[4] = q1;
      *(v4f*)&fv[8] = q2; *(v4f*)&fv[12] = q3;
      #pragma unroll
      for (int j = 0; j < 16; ++j) Bs[bn + j][bk] = f32_to_bf16(fv[j]);
    }
    __syncthreads();

    v16bf a0 = load_frag_a(&As[0][0], 40, wm * 32, lane);
    v16bf a1 = load_frag_a(&As[0][0], 40, wm * 32 + 16, lane);
    #pragma unroll
    for (int nj = 0; nj < 4; ++nj) {
      v16bf bb = load_frag_b(&Bs[0][0], 40, wn * 64 + nj * 16, 0, lane);
      acc[0][nj] = wmma_bf16(a0, bb, acc[0][nj]);
      acc[1][nj] = wmma_bf16(a1, bb, acc[1][nj]);
    }
    __syncthreads();
  }

  // epilogue: bias + scatter into per-head bf16 Q/K/V
  #pragma unroll
  for (int mi = 0; mi < 2; ++mi) {
    #pragma unroll
    for (int nj = 0; nj < 4; ++nj) {
      const int n = nBase + wn * 64 + nj * 16 + lo;
      const float bv = bias[n];
      const int tsel = n >> 10;          // 0=Q 1=K 2=V
      const int d = n & 1023;
      const int hh = d >> 6;             // head
      const int dk = d & 63;
      unsigned short* dst = (tsel == 0) ? Qb : (tsel == 1) ? Kb : Vb;
      #pragma unroll
      for (int v = 0; v < 8; ++v) {
        const int m = mBase + wm * 32 + mi * 16 + v + 8 * hi;
        const int bi = m >> 11, s = m & 2047;
        dst[(((size_t)(bi * 16 + hh)) * 2048 + s) * 64 + dk] =
            f32_to_bf16(acc[mi][nj][v] + bv);
      }
    }
  }
}

// ======================================================================
// Kernel 2: flash attention. One WG per (qblock64, head, batch).
// 4 waves; wave owns 16 query rows. Online softmax, causal mask.
// K tile staged with GLOBAL_LOAD_ASYNC_TO_LDS (pure bf16 copy).
// ======================================================================
__global__ __launch_bounds__(128) void attn_kernel(const unsigned short* __restrict__ Q,
                                                   const unsigned short* __restrict__ K,
                                                   const unsigned short* __restrict__ V,
                                                   unsigned short* __restrict__ O) {
  __shared__ __align__(16) unsigned short Ks[64][72];     // [key][dk]   (B-op for QK^T)
  __shared__ __align__(16) unsigned short Vs[64][72];     // [dk][key]   (B-op for PV)
  __shared__ __align__(16) unsigned short Ps[4][16][72];  // wave-private P tile [q][key]

  const int qblock = blockIdx.x;
  const int h = blockIdx.y;
  const int b = blockIdx.z;
  const size_t head_base = ((size_t)(b * 16 + h)) * 2048 * 64;

  const int t = threadIdx.x;
  const int lane = t & 31;
  const int wave = t >> 5;
  const int lo = lane & 15, hi = lane >> 4;
  const int qbase = qblock * 64 + wave * 16;

  // Q fragments straight from global bf16 (row-major == A-operand layout)
  const unsigned short* qptr = Q + head_base + (size_t)qbase * 64;
  const v16bf qa0 = load_frag_a(qptr, 64, 0, lane);        // dk 0..31
  const v16bf qa1 = load_frag_a(qptr + 32, 64, 0, lane);   // dk 32..63

  float mrow[8], lrow[8];
  v8f oacc[4] = {};
  #pragma unroll
  for (int v = 0; v < 8; ++v) { mrow[v] = -1e30f; lrow[v] = 0.0f; }

  const int kr = t >> 1;          // 0..63 key row to stage
  const int hb = (t & 1) * 32;    // 32-half chunk
  const unsigned ks_dst = lds_off32(&Ks[kr][hb]);

  for (int jb = 0; jb <= qblock; ++jb) {
    {  // K tile: async copy straight into LDS (no VGPR round-trip)
      const unsigned short* kg = K + head_base + (size_t)(jb * 64 + kr) * 64 + hb;
      async_copy64(ks_dst, kg);
      // V tile must be transposed -> manual staging
      const unsigned short* vg = V + head_base + (size_t)(jb * 64 + kr) * 64 + hb;
      union { v4u u[4]; unsigned short s[32]; } vt;
      const v4u* v4p = (const v4u*)vg;
      vt.u[0] = v4p[0]; vt.u[1] = v4p[1]; vt.u[2] = v4p[2]; vt.u[3] = v4p[3];
      #pragma unroll
      for (int i = 0; i < 32; ++i) Vs[hb + i][kr] = vt.s[i];   // transpose
      wait_async0();
    }
    __syncthreads();

    // S = Q K^T (16 x 64), two chained WMMAs over dk
    v8f sc[4];
    #pragma unroll
    for (int kb = 0; kb < 4; ++kb) {
      v16bf b0 = load_frag_b(&Ks[0][0], 72, kb * 16, 0, lane);
      v16bf b1 = load_frag_b(&Ks[0][0], 72, kb * 16, 32, lane);
      v8f c = {};
      c = wmma_bf16(qa0, b0, c);
      c = wmma_bf16(qa1, b1, c);
      sc[kb] = c;
    }
    // scale + causal mask  (D layout: col = lane&15, row = v + 8*(lane>>4))
    #pragma unroll
    for (int kb = 0; kb < 4; ++kb) {
      const int key = jb * 64 + kb * 16 + lo;
      #pragma unroll
      for (int v = 0; v < 8; ++v) {
        const int row = qbase + v + 8 * hi;
        const float x = sc[kb][v] * 0.125f;     // 1/sqrt(64)
        sc[kb][v] = (key > row) ? -1e30f : x;
      }
    }
    // online softmax (row reductions over 16-lane column groups)
    #pragma unroll
    for (int v = 0; v < 8; ++v) {
      float mx = fmaxf(fmaxf(sc[0][v], sc[1][v]), fmaxf(sc[2][v], sc[3][v]));
      mx = fmaxf(mx, __shfl_xor(mx, 1, 32));
      mx = fmaxf(mx, __shfl_xor(mx, 2, 32));
      mx = fmaxf(mx, __shfl_xor(mx, 4, 32));
      mx = fmaxf(mx, __shfl_xor(mx, 8, 32));
      const float mnew = fmaxf(mrow[v], mx);
      const float al = __expf(mrow[v] - mnew);
      mrow[v] = mnew;
      lrow[v] *= al;
      #pragma unroll
      for (int nt = 0; nt < 4; ++nt) oacc[nt][v] = oacc[nt][v] * al;
      float rs = 0.0f;
      #pragma unroll
      for (int kb = 0; kb < 4; ++kb) {
        const float p = __expf(sc[kb][v] - mnew);
        sc[kb][v] = p;
        rs += p;
      }
      rs += __shfl_xor(rs, 1, 32);
      rs += __shfl_xor(rs, 2, 32);
      rs += __shfl_xor(rs, 4, 32);
      rs += __shfl_xor(rs, 8, 32);
      lrow[v] += rs;
    }
    // D-layout -> A-operand layout via wave-private LDS tile (bf16)
    #pragma unroll
    for (int kb = 0; kb < 4; ++kb)
      #pragma unroll
      for (int v = 0; v < 8; ++v)
        Ps[wave][v + 8 * hi][kb * 16 + lo] = f32_to_bf16(sc[kb][v]);

    const v16bf p0 = load_frag_a(&Ps[wave][0][0], 72, 0, lane);    // keys 0..31
    const v16bf p1 = load_frag_a(&Ps[wave][0][32], 72, 0, lane);   // keys 32..63
    #pragma unroll
    for (int nt = 0; nt < 4; ++nt) {
      v16bf vb0 = load_frag_b(&Vs[0][0], 72, nt * 16, 0, lane);
      v16bf vb1 = load_frag_b(&Vs[0][0], 72, nt * 16, 32, lane);
      oacc[nt] = wmma_bf16(p0, vb0, oacc[nt]);
      oacc[nt] = wmma_bf16(p1, vb1, oacc[nt]);
    }
    __syncthreads();   // protect Ks/Vs before next block's staging
  }

  // normalize + store bf16 to O in [B,S,H*64] layout
  #pragma unroll
  for (int v = 0; v < 8; ++v) {
    const float inv = 1.0f / lrow[v];
    const int row = qbase + v + 8 * hi;
    #pragma unroll
    for (int nt = 0; nt < 4; ++nt) {
      const int dk = nt * 16 + lo;
      O[((size_t)(b * 2048 + row)) * 1024 + h * 64 + dk] =
          f32_to_bf16(oacc[nt][v] * inv);
    }
  }
}

// ======================================================================
// Kernel 3: out = O @ w_o + b_o  (M=8192, N=1024, K=1024), fp32 output
// A tile (already bf16) staged with GLOBAL_LOAD_ASYNC_TO_LDS.
// ======================================================================
__global__ __launch_bounds__(256) void out_gemm(const unsigned short* __restrict__ A16,
                                                const float* __restrict__ W,
                                                const float* __restrict__ bias,
                                                float* __restrict__ Out) {
  __shared__ __align__(16) unsigned short As[128][40];
  __shared__ __align__(16) unsigned short Bs[128][40];

  const int nBase = blockIdx.x * 128;
  const int mBase = blockIdx.y * 128;
  const int t = threadIdx.x;
  const int lane = t & 31;
  const int wave = t >> 5;
  const int wm = wave >> 1, wn = wave & 1;
  const int lo = lane & 15, hi = lane >> 4;

  v8f acc[2][4] = {};

  const int ar = t >> 1, ac = (t & 1) * 16;
  const int bk = t >> 3, bn = (t & 7) * 16;
  const unsigned as_dst = lds_off32(&As[ar][ac]);

  for (int k0 = 0; k0 < 1024; k0 += 32) {
    // A already bf16: async copy straight into LDS
    async_copy32(as_dst, A16 + (size_t)(mBase + ar) * 1024 + k0 + ac);
    {  // stage W transposed to Bs[n][k], f32 -> bf16
      const v4f* s4 = (const v4f*)(W + (size_t)(k0 + bk) * 1024 + nBase + bn);
      v4f q0 = s4[0], q1 = s4[1], q2 = s4[2], q3 = s4[3];
      float fv[16] __attribute__((aligned(16)));
      *(v4f*)&fv[0] = q0; *(v4f*)&fv[4] = q1;
      *(v4f*)&fv[8] = q2; *(v4f*)&fv[12] = q3;
      #pragma unroll
      for (int j = 0; j < 16; ++j) Bs[bn + j][bk] = f32_to_bf16(fv[j]);
    }
    wait_async0();
    __syncthreads();

    v16bf a0 = load_frag_a(&As[0][0], 40, wm * 32, lane);
    v16bf a1 = load_frag_a(&As[0][0], 40, wm * 32 + 16, lane);
    #pragma unroll
    for (int nj = 0; nj < 4; ++nj) {
      v16bf bb = load_frag_b(&Bs[0][0], 40, wn * 64 + nj * 16, 0, lane);
      acc[0][nj] = wmma_bf16(a0, bb, acc[0][nj]);
      acc[1][nj] = wmma_bf16(a1, bb, acc[1][nj]);
    }
    __syncthreads();
  }

  #pragma unroll
  for (int mi = 0; mi < 2; ++mi) {
    #pragma unroll
    for (int nj = 0; nj < 4; ++nj) {
      const int n = nBase + wn * 64 + nj * 16 + lo;
      const float bv = bias[n];
      #pragma unroll
      for (int v = 0; v < 8; ++v) {
        const int m = mBase + wm * 32 + mi * 16 + v + 8 * hi;
        Out[(size_t)m * 1024 + n] = acc[mi][nj][v] + bv;
      }
    }
  }
}

// ======================================================================
extern "C" void kernel_launch(void* const* d_in, const int* in_sizes, int n_in,
                              void* d_out, int out_size, void* d_ws, size_t ws_size,
                              hipStream_t stream) {
  (void)in_sizes; (void)n_in; (void)out_size; (void)ws_size;
  const float* x      = (const float*)d_in[0];
  // d_in[1] = causal mask (int32) — causality computed analytically, unused
  const float* w_qkv  = (const float*)d_in[2];
  const float* b_qkv  = (const float*)d_in[3];
  const float* w_o    = (const float*)d_in[4];
  const float* b_o    = (const float*)d_in[5];
  float* out = (float*)d_out;

  const size_t HSZ = (size_t)4 * 16 * 2048 * 64;   // 8M elems per tensor
  unsigned short* qb = (unsigned short*)d_ws;
  unsigned short* kb = qb + HSZ;
  unsigned short* vb = kb + HSZ;
  unsigned short* ob = vb + HSZ;                    // 64 MB total workspace

  qkv_gemm<<<dim3(3072 / 128, 8192 / 128), 256, 0, stream>>>(x, w_qkv, b_qkv, qb, kb, vb);
  attn_kernel<<<dim3(2048 / 64, 16, 4), 128, 0, stream>>>(qb, kb, vb, ob);
  out_gemm<<<dim3(1024 / 128, 8192 / 128), 256, 0, stream>>>(ob, w_o, b_o, out);
}
